// SparseConv2d_86208583565591
// MI455X (gfx1250) — matrix operand order repaired and checked by reading, not compiled, not run on hardware
//
#include <hip/hip_runtime.h>
#include <hip/hip_bf16.h>
#include <stdint.h>

// CDNA5 (gfx1250) wave32 WMMA types
typedef __attribute__((ext_vector_type(16))) __bf16 v16bf;
typedef __attribute__((ext_vector_type(8)))  __bf16 v8bf;
typedef __attribute__((ext_vector_type(8)))  float  v8f;

#define C_IN   64
#define C_OUT  128
#define H_IM   112
#define W_IM   112
#define B_N    16
#define KHW    9          // 3x3 taps
#define NT     16         // output-pixel tile (WMMA N dimension)
#define ROWS2  2          // output rows per block
#define LROWS  4          // LDS input rows = ROWS2 + 2 halo
#define COLS   18         // NT + 2 halo columns
#define CP     72         // LDS channel pitch in shorts (mult. of 8 -> 16B aligned chunks)
#define CPD    36         // CP/2 dwords

__device__ __forceinline__ unsigned short f2bf_rn(float f) {
  uint32_t u = __float_as_uint(f);
  uint32_t r = u + 0x7fffu + ((u >> 16) & 1u);   // round-to-nearest-even
  return (unsigned short)(r >> 16);
}
__device__ __forceinline__ float bf2f(unsigned short h) {
  return __uint_as_float(((uint32_t)h) << 16);
}

// ---------------------------------------------------------------------------
// Prep: spW = weight*mask, reorder [C_out][c*9+ij] -> [ij][o][c], split into
// bf16 hi/lo planes (bf16x3 decomposition of fp32).
// ---------------------------------------------------------------------------
__global__ void __launch_bounds__(256)
prep_weights(const float* __restrict__ weight, const float* __restrict__ mask,
             unsigned short* __restrict__ wHi, unsigned short* __restrict__ wLo) {
  int t = blockIdx.x * 256 + threadIdx.x;
  const int total = KHW * C_OUT * C_IN;
  if (t >= total) return;
  int c  = t & 63;
  int o  = (t >> 6) & 127;
  int ij = t >> 13;                       // 0..8
  int k  = c * KHW + ij;                  // reference K ordering: (c,i,j)
  float w = weight[o * (C_IN * KHW) + k] * mask[o * (C_IN * KHW) + k];
  unsigned short hi = f2bf_rn(w);
  unsigned short lo = f2bf_rn(w - bf2f(hi));
  wHi[t] = hi;
  wLo[t] = lo;
}

// ---------------------------------------------------------------------------
// Main: implicit-im2col GEMM, M=128 (C_out), N=16 pixels x 2 output rows per
// block. K = 9 taps x 64 channels; v_wmma_f32_16x16x32_bf16, bf16x3 split.
// Block = 256 threads = 8 waves; wave w owns C_out rows [16w, 16w+16) and
// carries two accumulators (output rows h0, h0+1). B fragments for LDS rows
// 0..3 are loaded once per (j, c-block) and shared across taps/rows.
// ---------------------------------------------------------------------------
__global__ void __launch_bounds__(256)
sparse_conv_wmma(const float* __restrict__ x,
                 const unsigned short* __restrict__ wHi,
                 const unsigned short* __restrict__ wLo,
                 const float* __restrict__ bias,
                 float* __restrict__ out) {
  __shared__ unsigned short sHi[LROWS * COLS * CP];
  __shared__ unsigned short sLo[LROWS * COLS * CP];

  const int tid  = threadIdx.x;
  const int lane = tid & 31;
  const int wv   = tid >> 5;              // wave id 0..7 -> M tile

  const int blk = blockIdx.x;
  const int wt  = (blk % (W_IM / NT)) * NT;                   // pixel base
  const int h0  = ((blk / (W_IM / NT)) % (H_IM / ROWS2)) * ROWS2; // row pair
  const int b   = blk / ((W_IM / NT) * (H_IM / ROWS2));       // batch

  // ---- Stage x slab: rows h0-1..h0+2, cols wt-1..wt+16, all 64 channels.
  // Packed bf16 pairs (2 channels / dword): conflict-free ds_store_b32.
  // 4*18*32 = 2304 pairs = exactly 9 iterations for all 256 threads.
  for (int p = tid; p < LROWS * COLS * 32; p += 256) {
    int cd  = p & 31;                 // channel-pair index (c = 2*cd)
    int sp  = p >> 5;                 // r*COLS + col
    int col = sp % COLS;
    int r   = sp / COLS;
    int hin = h0 + r - 1;
    int win = wt + col - 1;
    float f0 = 0.0f, f1 = 0.0f;
    if ((unsigned)hin < (unsigned)H_IM && (unsigned)win < (unsigned)W_IM) {
      const float* xp = x + (((size_t)b * C_IN + 2 * cd) * H_IM + hin) * W_IM + win;
      f0 = xp[0];
      f1 = xp[(size_t)H_IM * W_IM];
    }
    unsigned short h0b = f2bf_rn(f0), h1b = f2bf_rn(f1);
    unsigned short l0b = f2bf_rn(f0 - bf2f(h0b)), l1b = f2bf_rn(f1 - bf2f(h1b));
    const int di = sp * CPD + cd;
    ((uint32_t*)sHi)[di] = (uint32_t)h0b | ((uint32_t)h1b << 16);
    ((uint32_t*)sLo)[di] = (uint32_t)l0b | ((uint32_t)l1b << 16);
  }
  __syncthreads();

  // ---- K loop ----
  v8f acc0 = {0.f, 0.f, 0.f, 0.f, 0.f, 0.f, 0.f, 0.f};
  v8f acc1 = {0.f, 0.f, 0.f, 0.f, 0.f, 0.f, 0.f, 0.f};
  const int n    = lane & 15;                    // pixel within tile / N col
  const int half = (lane >> 4) & 1;
  const int asel = half * 16;                    // A frag: K 0-15 vs 16-31
  const int bsel = half * 16;                    // B frag: K 0-15 vs 16-31
  const int mrow = wv * 16 + n;                  // A matrix row (C_out index)

#pragma unroll
  for (int j = 0; j < 3; ++j) {
#pragma unroll
    for (int cb = 0; cb < 2; ++cb) {
      const int c0 = cb * 32;
      // Load B fragments for all 4 LDS rows once; row r serves output row 0
      // at tap i=r and output row 1 at tap i=r-1.
      v16bf bh[LROWS], bl[LROWS];
#pragma unroll
      for (int r = 0; r < LROWS; ++r) {
        const unsigned short* bH = &sHi[(r * COLS + n + j) * CP + c0 + bsel];
        const unsigned short* bL = &sLo[(r * COLS + n + j) * CP + c0 + bsel];
        ((v8bf*)&bh[r])[0] = *(const v8bf*)(bH);
        ((v8bf*)&bh[r])[1] = *(const v8bf*)(bH + 8);
        ((v8bf*)&bl[r])[0] = *(const v8bf*)(bL);
        ((v8bf*)&bl[r])[1] = *(const v8bf*)(bL + 8);
      }
#pragma unroll
      for (int i = 0; i < 3; ++i) {
        const int ij = i * 3 + j;
        const unsigned short* aH = wHi + (size_t)(ij * C_OUT + mrow) * C_IN + c0 + asel;
        const unsigned short* aL = wLo + (size_t)(ij * C_OUT + mrow) * C_IN + c0 + asel;
        v16bf a_hi, a_lo;
        ((v8bf*)&a_hi)[0] = *(const v8bf*)(aH);
        ((v8bf*)&a_hi)[1] = *(const v8bf*)(aH + 8);
        ((v8bf*)&a_lo)[0] = *(const v8bf*)(aL);
        ((v8bf*)&a_lo)[1] = *(const v8bf*)(aL + 8);
        // bf16x3: hi*hi + hi*lo + lo*hi, f32 accumulate; two output rows
        acc0 = __builtin_amdgcn_wmma_f32_16x16x32_bf16(false, a_hi, false, bh[i],
                                                       (short)0, acc0, false, false);
        acc1 = __builtin_amdgcn_wmma_f32_16x16x32_bf16(false, a_hi, false, bh[i + 1],
                                                       (short)0, acc1, false, false);
        acc0 = __builtin_amdgcn_wmma_f32_16x16x32_bf16(false, a_hi, false, bl[i],
                                                       (short)0, acc0, false, false);
        acc1 = __builtin_amdgcn_wmma_f32_16x16x32_bf16(false, a_hi, false, bl[i + 1],
                                                       (short)0, acc1, false, false);
        acc0 = __builtin_amdgcn_wmma_f32_16x16x32_bf16(false, a_lo, false, bh[i],
                                                       (short)0, acc0, false, false);
        acc1 = __builtin_amdgcn_wmma_f32_16x16x32_bf16(false, a_lo, false, bh[i + 1],
                                                       (short)0, acc1, false, false);
      }
    }
  }

  // ---- Epilogue: C/D layout — VGPR g: M = base+g (lanes 0-15) / base+8+g.
  // Output is streamed (never re-read): non-temporal stores keep x/weights
  // resident in L2.
  const int obase = wv * 16 + half * 8;
#pragma unroll
  for (int g = 0; g < 8; ++g) {
    const int o = obase + g;
    const float bv = bias[o];
    float* op = out + (((size_t)b * C_OUT + o) * H_IM + h0) * W_IM + wt + n;
    __builtin_nontemporal_store(acc0[g] + bv, op);
    __builtin_nontemporal_store(acc1[g] + bv, op + W_IM);
  }
}

// ---------------------------------------------------------------------------
extern "C" void kernel_launch(void* const* d_in, const int* in_sizes, int n_in,
                              void* d_out, int out_size, void* d_ws, size_t ws_size,
                              hipStream_t stream) {
  const float* x      = (const float*)d_in[0];
  const float* weight = (const float*)d_in[1];
  const float* mask   = (const float*)d_in[2];
  const float* bias   = (const float*)d_in[3];
  float* out = (float*)d_out;

  // Workspace: bf16 hi/lo weight planes, [9][128][64] each -> 288 KB total.
  unsigned short* wHi = (unsigned short*)d_ws;
  unsigned short* wLo = wHi + (size_t)KHW * C_OUT * C_IN;

  const int totalW = KHW * C_OUT * C_IN;
  prep_weights<<<(totalW + 255) / 256, 256, 0, stream>>>(weight, mask, wHi, wLo);

  const int nblk = B_N * (H_IM / ROWS2) * (W_IM / NT);   // 16*56*7 = 6272 tiles
  sparse_conv_wmma<<<nblk, 256, 0, stream>>>(x, wHi, wLo, bias, out);
}
